// MambaBlock_25580825215307
// MI455X (gfx1250) — matrix operand order repaired and checked by reading, not compiled, not run on hardware
//
#include <hip/hip_runtime.h>

// ---------------- problem constants ----------------
#define DM   1024            // D_MODEL
#define DS   16              // D_STATE
#define BN   8               // BATCH
#define LSEQ 2048            // SEQ
#define MTOT (BN*LSEQ)       // 16384 rows
#define NCH  8               // scan chunks per sequence
#define CLEN (LSEQ/NCH)      // 256 steps per chunk

typedef __attribute__((ext_vector_type(16))) _Float16 v16h;
typedef __attribute__((ext_vector_type(8)))  float    v8f;
typedef __attribute__((ext_vector_type(4)))  _Float16 v4h;

union FragH { uint4 u4[2]; v16h v; };

__device__ __forceinline__ float sigm(float z) { return 1.0f / (1.0f + __expf(-z)); }

// ---- gfx1250 async global->LDS copy (ASYNCcnt-tracked), 16 bytes per lane ----
__device__ __forceinline__ void async_copy_b128(unsigned lds_byte_addr, const _Float16* gptr) {
  asm volatile("global_load_async_to_lds_b128 %0, %1, off"
               :: "v"(lds_byte_addr), "v"(gptr)
               : "memory");
}
__device__ __forceinline__ void wait_async_le2() {
  asm volatile("s_wait_asynccnt 0x2" ::: "memory");
}
__device__ __forceinline__ void wait_async_le0() {
  asm volatile("s_wait_asynccnt 0x0" ::: "memory");
}

// ---------------- f32 -> f16 conversion, 4 elems/thread ----------------
__global__ __launch_bounds__(256) void cvt_f32_f16_x4(const float* __restrict__ in,
                                                      _Float16* __restrict__ out, int n4) {
  int i = blockIdx.x * 256 + threadIdx.x;
  if (i >= n4) return;
  float4 f = ((const float4*)in)[i];
  v4h h;
  h.x = (_Float16)f.x; h.y = (_Float16)f.y; h.z = (_Float16)f.z; h.w = (_Float16)f.w;
  ((v4h*)out)[i] = h;
}

// ---------------- WMMA GEMM:  out[M,N] = A[M,K] * W[N,K]^T + bias ----------------
// block = 256 threads = 8 waves; block tile 128(M) x 128(N); wave tile 64 x 32.
// A tile (128x32 f16, 8KB) is double-buffered in LDS via async global->LDS loads;
// B (weights) is read directly from global (L2-resident, reused by every M block).
template <bool OUT_F32>
__global__ __launch_bounds__(256) void wmma_gemm_f16(const _Float16* __restrict__ A,
                                                     const _Float16* __restrict__ W,
                                                     const float*   __restrict__ bias,
                                                     void* __restrict__ outv,
                                                     int M, int N, int K) {
  __shared__ _Float16 sA[2][128 * 32];   // 2 x 8KB double buffer

  const int t    = threadIdx.x;
  const int lane = t & 31;
  const int wave = t >> 5;
  const int wm   = wave >> 2;       // 0..1
  const int wn   = wave & 3;        // 0..3
  const int row  = lane & 15;       // M-row (A) / N-col (B) within 16
  const int hs   = lane >> 4;       // which K-half this lane carries

  const int mblk = blockIdx.y * 128;
  const int m0   = mblk + wm * 64;
  const int n0   = blockIdx.x * 128 + wn * 32;

  // --- async staging addressing: thread t copies 32B of the 8KB tile ---
  // tile row = t/2 (each row = 32 halfs = 64B, covered by two threads)
  const int ar = t >> 1;             // 0..127
  const int ac = (t & 1) * 16;       // half-offset within row
  const _Float16* Ag = A + (size_t)(mblk + ar) * K + ac;
  const unsigned ldsA0 = (unsigned)(uintptr_t)&sA[0][ar * 32 + ac];
  const unsigned ldsA1 = (unsigned)(uintptr_t)&sA[1][ar * 32 + ac];

  v8f acc[4][2];
  v8f zero = {0.f,0.f,0.f,0.f,0.f,0.f,0.f,0.f};
#pragma unroll
  for (int i = 0; i < 4; ++i)
#pragma unroll
    for (int j = 0; j < 2; ++j) acc[i][j] = zero;

  const _Float16* Wbase = W + (size_t)(n0 + row) * K;
  const int KT = K / 32;

  // prologue: stage k-tile 0 into buffer 0 (2 async-load instructions per wave)
  async_copy_b128(ldsA0,      Ag);
  async_copy_b128(ldsA0 + 16, Ag + 8);

  for (int ki = 0; ki < KT; ++ki) {
    const int cur = ki & 1;
    const int k   = ki * 32;

    if (ki + 1 < KT) {   // prefetch next k-tile into the other buffer
      const unsigned ldst = cur ? ldsA0 : ldsA1;
      const _Float16* gp  = Ag + (size_t)(k + 32);
      async_copy_b128(ldst,      gp);
      async_copy_b128(ldst + 16, gp + 8);
      wait_async_le2();          // current tile done (async loads complete in order)
    } else {
      wait_async_le0();
    }
    __syncthreads();             // current tile visible to all 8 waves

    FragH a[4], b[2];
    // A 16x32 f16 fragments from LDS: lanes 0-15 take K [0,8)+[16,24), lanes 16-31 the rest
#pragma unroll
    for (int mt = 0; mt < 4; ++mt) {
      const int r = wm * 64 + mt * 16 + row;
      a[mt].u4[0] = *(const uint4*)&sA[cur][r * 32 + hs * 8];
      a[mt].u4[1] = *(const uint4*)&sA[cur][r * 32 + 16 + hs * 8];
    }
    // B 32x16 f16: lanes 0-15 take K [0,8)+[16,24), lanes 16-31 the rest
#pragma unroll
    for (int nt = 0; nt < 2; ++nt) {
      const _Float16* p = Wbase + (size_t)nt * 16 * K + k + hs * 8;
      b[nt].u4[0] = *(const uint4*)(p);
      b[nt].u4[1] = *(const uint4*)(p + 16);
    }
#pragma unroll
    for (int mt = 0; mt < 4; ++mt)
#pragma unroll
      for (int nt = 0; nt < 2; ++nt)
        acc[mt][nt] = __builtin_amdgcn_wmma_f32_16x16x32_f16(
            false, a[mt].v, false, b[nt].v, (short)0, acc[mt][nt], false, false);

    __syncthreads();             // all waves done with this buffer before it is refilled
  }

  // C/D layout: vgpr r at lane -> (m = 8*hs + r, n = row) within the 16x16 tile
#pragma unroll
  for (int mt = 0; mt < 4; ++mt) {
#pragma unroll
    for (int nt = 0; nt < 2; ++nt) {
      const int n  = n0 + nt * 16 + row;
      const float bv = bias[n];
      const int mb = m0 + mt * 16 + hs * 8;
#pragma unroll
      for (int r = 0; r < 8; ++r) {
        float val = acc[mt][nt][r] + bv;
        if (OUT_F32) ((float*)outv)[(size_t)(mb + r) * N + n] = val;
        else ((_Float16*)outv)[(size_t)(mb + r) * N + n] = (_Float16)val;
      }
    }
  }
}

// ---------------- chunked conv+SiLU+SSM scan ----------------
// hid layout: [B*L, 2*DM] f16; columns [0,DM) = x1, [DM,2DM) = v_pre.

// pass1: per-chunk partial state (h starts at 0), state only (no output).
__global__ __launch_bounds__(256) void scan_pass1(const _Float16* __restrict__ hid,
                                                  const float* __restrict__ convw,
                                                  const float* __restrict__ Aarr,
                                                  const float* __restrict__ Bsv,
                                                  float* __restrict__ hend) {
  int g = blockIdx.x * 256 + threadIdx.x;       // (b*NCH + j)*DM + c
  int c = g % DM;
  int j = (g / DM) % NCH;
  int b = g / (DM * NCH);

  float Ab[DS], Bv[DS], h[DS];
#pragma unroll
  for (int s = 0; s < DS; ++s) {
    Ab[s] = sigm(Aarr[c * DS + s]);
    Bv[s] = Bsv[c * DS + s];
    h[s]  = 0.f;
  }
  float w0 = convw[c * 3 + 0], w1 = convw[c * 3 + 1], w2 = convw[c * 3 + 2];

  const size_t strd = 2 * DM;
  const _Float16* xp = hid + (size_t)b * LSEQ * strd + c;
  const int l0 = j * CLEN;
  float xm1 = (l0 > 0) ? (float)xp[(size_t)(l0 - 1) * strd] : 0.f;
  float x0  = (float)xp[(size_t)l0 * strd];
  for (int t = 0; t < CLEN; ++t) {
    int l = l0 + t;
    float xpn = (l + 1 < LSEQ) ? (float)xp[(size_t)(l + 1) * strd] : 0.f;
    float xc  = w0 * xm1 + w1 * x0 + w2 * xpn;
    float sc  = xc * sigm(xc);
#pragma unroll
    for (int s = 0; s < DS; ++s) h[s] = h[s] * Ab[s] + sc * Bv[s];
    xm1 = x0; x0 = xpn;
  }
  float* hd = hend + ((size_t)(b * DM + c) * NCH + j) * DS;
#pragma unroll
  for (int s = 0; s < DS; ++s) hd[s] = h[s];
}

// pass2: serial combine across the 8 chunks: h0_j = Abar^CLEN * h0_{j-1} + hend_{j-1}
__global__ __launch_bounds__(256) void scan_combine(const float* __restrict__ Aarr,
                                                    const float* __restrict__ hend,
                                                    float* __restrict__ h0) {
  int g = blockIdx.x * 256 + threadIdx.x;       // b*DM + c
  int c = g % DM;
  float AT[DS];
#pragma unroll
  for (int s = 0; s < DS; ++s) {
    float a = sigm(Aarr[c * DS + s]);
#pragma unroll
    for (int q = 0; q < 8; ++q) a = a * a;      // a^256 (CLEN = 2^8)
    AT[s] = a;
  }
  const float* he = hend + (size_t)g * NCH * DS;
  float* hz = h0 + (size_t)g * NCH * DS;
  float h[DS];
#pragma unroll
  for (int s = 0; s < DS; ++s) h[s] = 0.f;
  for (int j = 0; j < NCH; ++j) {
#pragma unroll
    for (int s = 0; s < DS; ++s) hz[j * DS + s] = h[s];
#pragma unroll
    for (int s = 0; s < DS; ++s) h[s] = AT[s] * h[s] + he[j * DS + s];
  }
}

// pass3: full scan per chunk with correct initial state; fuse y = C.h + D*x and SiLU gate.
__global__ __launch_bounds__(256) void scan_pass3(const _Float16* __restrict__ hid,
                                                  const float* __restrict__ convw,
                                                  const float* __restrict__ Aarr,
                                                  const float* __restrict__ Bsv,
                                                  const float* __restrict__ Csv,
                                                  const float* __restrict__ Dsv,
                                                  const float* __restrict__ h0,
                                                  _Float16* __restrict__ u) {
  int g = blockIdx.x * 256 + threadIdx.x;       // (b*NCH + j)*DM + c
  int c = g % DM;
  int j = (g / DM) % NCH;
  int b = g / (DM * NCH);

  float Ab[DS], Bv[DS], Cv[DS], h[DS];
  const float* hz = h0 + ((size_t)(b * DM + c) * NCH + j) * DS;
#pragma unroll
  for (int s = 0; s < DS; ++s) {
    Ab[s] = sigm(Aarr[c * DS + s]);
    Bv[s] = Bsv[c * DS + s];
    Cv[s] = Csv[c * DS + s];
    h[s]  = hz[s];
  }
  float w0 = convw[c * 3 + 0], w1 = convw[c * 3 + 1], w2 = convw[c * 3 + 2];
  float Dv = Dsv[c];

  const size_t strd = 2 * DM;
  const _Float16* xp = hid + (size_t)b * LSEQ * strd + c;
  const int l0 = j * CLEN;
  float xm1 = (l0 > 0) ? (float)xp[(size_t)(l0 - 1) * strd] : 0.f;
  float x0  = (float)xp[(size_t)l0 * strd];
  for (int t = 0; t < CLEN; ++t) {
    int l = l0 + t;
    float xpn = (l + 1 < LSEQ) ? (float)xp[(size_t)(l + 1) * strd] : 0.f;
    float xc  = w0 * xm1 + w1 * x0 + w2 * xpn;
    float sc  = xc * sigm(xc);
    float y   = Dv * sc;
#pragma unroll
    for (int s = 0; s < DS; ++s) {
      h[s] = h[s] * Ab[s] + sc * Bv[s];
      y += h[s] * Cv[s];
    }
    float v = (float)xp[(size_t)l * strd + DM];   // v_pre
    float gate = v * sigm(v);
    u[(size_t)(b * LSEQ + l) * DM + c] = (_Float16)(y * gate);
    xm1 = x0; x0 = xpn;
  }
}

// ---------------- launcher ----------------
extern "C" void kernel_launch(void* const* d_in, const int* in_sizes, int n_in,
                              void* d_out, int out_size, void* d_ws, size_t ws_size,
                              hipStream_t stream) {
  (void)in_sizes; (void)n_in; (void)out_size; (void)ws_size;
  const float* x     = (const float*)d_in[0];
  const float* w1w   = (const float*)d_in[1];
  const float* w1b   = (const float*)d_in[2];
  const float* v1w   = (const float*)d_in[3];
  const float* v1b   = (const float*)d_in[4];
  const float* w2w   = (const float*)d_in[5];
  const float* w2b   = (const float*)d_in[6];
  const float* convw = (const float*)d_in[7];
  const float* Aarr  = (const float*)d_in[8];
  const float* Bssm  = (const float*)d_in[9];
  const float* Cssm  = (const float*)d_in[10];
  const float* Dssm  = (const float*)d_in[11];
  float* out = (float*)d_out;

  // workspace layout (all offsets 256B-aligned); total ~149 MB
  char* ws = (char*)d_ws;
  _Float16* xb   = (_Float16*)(ws + 0);           // 32 MB  x as f16 [16384,1024]
  _Float16* wpk  = (_Float16*)(ws + 33554432ull); //  4 MB  packed [w1;v1] f16 [2048,1024]
  _Float16* w2h  = (_Float16*)(ws + 37748736ull); //  2 MB  w2 f16 [1024,1024]
  float*    bpk  = (float*)   (ws + 39845888ull); //  8 KB  packed [w1_b;v1_b]
  _Float16* hid  = (_Float16*)(ws + 39854080ull); // 64 MB  [16384,2048] f16 (x1 | v_pre)
  _Float16* ub   = (_Float16*)(ws + 106962944ull);// 32 MB  gated scan output f16
  float*    hend = (float*)   (ws + 140517376ull);//  4 MB  chunk-final states
  float*    h0   = (float*)   (ws + 144711680ull);//  4 MB  chunk-initial states

  // 1) convert operands to f16
  {
    int n4 = MTOT * DM / 4;
    cvt_f32_f16_x4<<<(n4 + 255) / 256, 256, 0, stream>>>(x, xb, n4);
    int w4 = DM * DM / 4;
    cvt_f32_f16_x4<<<(w4 + 255) / 256, 256, 0, stream>>>(w1w, wpk, w4);
    cvt_f32_f16_x4<<<(w4 + 255) / 256, 256, 0, stream>>>(v1w, wpk + (size_t)DM * DM, w4);
    cvt_f32_f16_x4<<<(w4 + 255) / 256, 256, 0, stream>>>(w2w, w2h, w4);
  }
  hipMemcpyAsync(bpk,        w1b, DM * sizeof(float), hipMemcpyDeviceToDevice, stream);
  hipMemcpyAsync(bpk + DM,   v1b, DM * sizeof(float), hipMemcpyDeviceToDevice, stream);

  // 2) fused input GEMM: hid = x @ [w1;v1]^T + [b1;bv]   (f16 out)
  {
    dim3 grid(2 * DM / 128, MTOT / 128);  // (16,128)
    wmma_gemm_f16<false><<<grid, 256, 0, stream>>>(xb, wpk, bpk, hid, MTOT, 2 * DM, DM);
  }

  // 3) chunked conv + SiLU + SSM + gate
  scan_pass1 <<<(BN * NCH * DM) / 256, 256, 0, stream>>>(hid, convw, Aarr, Bssm, hend);
  scan_combine<<<(BN * DM) / 256,      256, 0, stream>>>(Aarr, hend, h0);
  scan_pass3 <<<(BN * NCH * DM) / 256, 256, 0, stream>>>(hid, convw, Aarr, Bssm, Cssm, Dssm, h0, ub);

  // 4) output GEMM: out = u @ w2^T + b2   (f32 out)
  {
    dim3 grid(DM / 128, MTOT / 128);      // (8,128)
    wmma_gemm_f16<true><<<grid, 256, 0, stream>>>(ub, w2h, w2b, out, MTOT, DM, DM);
  }
}
